// MultiLayerGRU_10273561772336
// MI455X (gfx1250) — matrix operand order repaired and checked by reading, not compiled, not run on hardware
//
#include <hip/hip_runtime.h>

#define B_ 256
#define S_ 256
#define IN_ 16
#define H_ 512
#define H3_ 1536
#define L_ 3

typedef __attribute__((ext_vector_type(16))) __bf16 v16bf;
typedef __attribute__((ext_vector_type(8)))  __bf16 v8bf;
typedef __attribute__((ext_vector_type(8)))  float  v8f;

union V16U { v16bf v; v8bf h[2]; };

// A-matrix 16x32 bf16 fragment (contiguous k order, same as the B loads):
// lane m = lane&15 holds row rowbase+m; element e -> k = 16*half + e
// -> one contiguous 16-element run per lane (two 16B loads).
__device__ __forceinline__ v16bf load_a16x32(const __bf16* __restrict__ rowbase,
                                             int astride, int k0, int m, int half) {
  V16U u;
  const __bf16* p = rowbase + (size_t)m * astride + k0 + 16 * half;
  u.h[0] = *(const v8bf*)p;
  u.h[1] = *(const v8bf*)(p + 8);
  return u.v;
}

// Fast activations: v_exp_f32 + v_rcp_f32 (bf16-quantized pipeline, IEEE
// division is wasted effort here).
__device__ __forceinline__ float fast_sigmoid(float v) {
  return __builtin_amdgcn_rcpf(1.0f + __expf(-v));
}
__device__ __forceinline__ float fast_tanh(float v) {
  // tanh(x) = 2/(1+e^-2x) - 1
  return 2.0f * __builtin_amdgcn_rcpf(1.0f + __expf(-2.0f * v)) - 1.0f;
}

// ---------------- prep kernels ----------------

// W: K x N row-major (f32) -> Wt: N x Kpad col-major bf16 (zero-padded K)
__global__ void transpose_w(const float* __restrict__ W, __bf16* __restrict__ Wt,
                            int K, int N, int Kpad) {
  int total = N * Kpad;
  for (int idx = blockIdx.x * blockDim.x + threadIdx.x; idx < total;
       idx += gridDim.x * blockDim.x) {
    int n = idx / Kpad;
    int k = idx - n * Kpad;
    float v = (k < K) ? W[(size_t)k * N + n] : 0.0f;
    Wt[idx] = (__bf16)v;
  }
}

// xin[(b*S + t)*32 + k] = [timestamp | x(16) | zeros(15)] as bf16
__global__ void pack_xin(const float* __restrict__ x, const float* __restrict__ ts,
                         __bf16* __restrict__ xin) {
  const int total = B_ * S_ * 32;
  for (int i = blockIdx.x * blockDim.x + threadIdx.x; i < total;
       i += gridDim.x * blockDim.x) {
    int bt = i >> 5;
    int k = i & 31;
    float v = 0.0f;
    if (k == 0)
      v = ts[bt];
    else if (k <= IN_)
      v = x[(size_t)bt * IN_ + (k - 1)];
    xin[i] = (__bf16)v;
  }
}

__global__ void zero_h(float* __restrict__ hf, __bf16* __restrict__ hb, int n) {
  for (int i = blockIdx.x * blockDim.x + threadIdx.x; i < n;
       i += gridDim.x * blockDim.x) {
    hf[i] = 0.0f;
    hb[i] = (__bf16)0.0f;
  }
}

__global__ void copy_final(const float* __restrict__ hf, float* __restrict__ dst) {
  const int BH = B_ * H_;
  const int total = L_ * BH;
  for (int i = blockIdx.x * blockDim.x + threadIdx.x; i < total;
       i += gridDim.x * blockDim.x) {
    int l = i / BH;
    int rem = i - l * BH;
    dst[i] = hf[((size_t)l * 2 + 0) * BH + rem];  // final state lives in buf 0
  }
}

// ---------------- GRU step kernel ----------------
// One wave per 32x16 output tile (2 row-blocks x 1 col-block); B fragments
// reused across the two row-blocks. grid = 64 blocks x 128 threads (4 waves).
// cb = blockIdx>>1 (32 col blocks of H), rsb = (blockIdx&1)*4 + wave.
// KIN: input-GEMM depth (32 for layer 0 via packed xin, 512 otherwise);
// also the column stride of WiT.
template <int KIN>
__global__ __launch_bounds__(128) void gru_step(
    const __bf16* __restrict__ inA, int astride,
    const __bf16* __restrict__ WiT, const float* __restrict__ bi,
    const __bf16* __restrict__ WhT, const float* __restrict__ bh,
    const __bf16* __restrict__ hprev_bf16, const float* __restrict__ hprev_f32,
    __bf16* __restrict__ hnew_bf16, float* __restrict__ hnew_f32,
    float* __restrict__ outp, int t) {
  const int lane = threadIdx.x & 31;
  const int wave = threadIdx.x >> 5;
  const int cb = blockIdx.x >> 1;
  const int rsb = ((blockIdx.x & 1) << 2) | wave;
  const int m = lane & 15;
  const int half = lane >> 4;
  const int nl = lane & 15;
  const int r0 = rsb * 32;        // 32 rows per wave (two 16-row A tiles)
  const int c0 = cb * 16;
  const int colR = c0 + nl;
  const int colZ = H_ + c0 + nl;
  const int colN = 2 * H_ + c0 + nl;

  v8f acc_r[2], acc_z[2], acc_in[2], acc_hn[2];
  {
    float vr = bi[colR] + bh[colR];
    float vz = bi[colZ] + bh[colZ];
    float vi = bi[colN];
    float vh = bh[colN];
#pragma unroll
    for (int s = 0; s < 2; ++s)
#pragma unroll
      for (int j = 0; j < 8; ++j) {
        acc_r[s][j] = vr;
        acc_z[s][j] = vz;
        acc_in[s][j] = vi;
        acc_hn[s][j] = vh;
      }
  }

  // ---- input-side GEMM: gi = xin @ Wi (+bi already in acc) ----
  {
    const __bf16* arow0 = inA + (size_t)r0 * astride;
    const __bf16* arow1 = arow0 + (size_t)16 * astride;
#pragma unroll 1
    for (int k0 = 0; k0 < KIN; k0 += 32) {
      v16bf a0 = load_a16x32(arow0, astride, k0, m, half);
      v16bf a1 = load_a16x32(arow1, astride, k0, m, half);
      v16bf bR = *(const v16bf*)(WiT + (size_t)colR * KIN + k0 + 16 * half);
      v16bf bZ = *(const v16bf*)(WiT + (size_t)colZ * KIN + k0 + 16 * half);
      v16bf bN = *(const v16bf*)(WiT + (size_t)colN * KIN + k0 + 16 * half);
      acc_r[0]  = __builtin_amdgcn_wmma_f32_16x16x32_bf16(false, a0, false, bR, (short)0, acc_r[0],  false, false);
      acc_r[1]  = __builtin_amdgcn_wmma_f32_16x16x32_bf16(false, a1, false, bR, (short)0, acc_r[1],  false, false);
      acc_z[0]  = __builtin_amdgcn_wmma_f32_16x16x32_bf16(false, a0, false, bZ, (short)0, acc_z[0],  false, false);
      acc_z[1]  = __builtin_amdgcn_wmma_f32_16x16x32_bf16(false, a1, false, bZ, (short)0, acc_z[1],  false, false);
      acc_in[0] = __builtin_amdgcn_wmma_f32_16x16x32_bf16(false, a0, false, bN, (short)0, acc_in[0], false, false);
      acc_in[1] = __builtin_amdgcn_wmma_f32_16x16x32_bf16(false, a1, false, bN, (short)0, acc_in[1], false, false);
    }
  }

  // ---- hidden-side GEMM: gh = h_prev @ Wh (+bh already in acc) ----
  {
    const __bf16* arow0 = hprev_bf16 + (size_t)r0 * H_;
    const __bf16* arow1 = arow0 + (size_t)16 * H_;
#pragma unroll 1
    for (int k0 = 0; k0 < H_; k0 += 32) {
      v16bf a0 = load_a16x32(arow0, H_, k0, m, half);
      v16bf a1 = load_a16x32(arow1, H_, k0, m, half);
      v16bf bR = *(const v16bf*)(WhT + (size_t)colR * H_ + k0 + 16 * half);
      v16bf bZ = *(const v16bf*)(WhT + (size_t)colZ * H_ + k0 + 16 * half);
      v16bf bN = *(const v16bf*)(WhT + (size_t)colN * H_ + k0 + 16 * half);
      acc_r[0]  = __builtin_amdgcn_wmma_f32_16x16x32_bf16(false, a0, false, bR, (short)0, acc_r[0],  false, false);
      acc_r[1]  = __builtin_amdgcn_wmma_f32_16x16x32_bf16(false, a1, false, bR, (short)0, acc_r[1],  false, false);
      acc_z[0]  = __builtin_amdgcn_wmma_f32_16x16x32_bf16(false, a0, false, bZ, (short)0, acc_z[0],  false, false);
      acc_z[1]  = __builtin_amdgcn_wmma_f32_16x16x32_bf16(false, a1, false, bZ, (short)0, acc_z[1],  false, false);
      acc_hn[0] = __builtin_amdgcn_wmma_f32_16x16x32_bf16(false, a0, false, bN, (short)0, acc_hn[0], false, false);
      acc_hn[1] = __builtin_amdgcn_wmma_f32_16x16x32_bf16(false, a1, false, bN, (short)0, acc_hn[1], false, false);
    }
  }

  // ---- gates + state update. C/D layout: row = j + 8*half, col = lane&15 ----
  const int colH = c0 + nl;
#pragma unroll
  for (int s = 0; s < 2; ++s) {
#pragma unroll
    for (int j = 0; j < 8; ++j) {
      int row = r0 + s * 16 + j + 8 * half;
      float r = fast_sigmoid(acc_r[s][j]);
      float z = fast_sigmoid(acc_z[s][j]);
      float nn = fast_tanh(acc_in[s][j] + r * acc_hn[s][j]);
      float hp = hprev_f32[(size_t)row * H_ + colH];
      float hnew = (1.0f - z) * nn + z * hp;
      hnew_f32[(size_t)row * H_ + colH] = hnew;
      hnew_bf16[(size_t)row * H_ + colH] = (__bf16)hnew;
      if (outp) outp[(size_t)row * S_ * H_ + (size_t)t * H_ + colH] = hnew;
    }
  }
}

// ---------------- host ----------------

extern "C" void kernel_launch(void* const* d_in, const int* in_sizes, int n_in,
                              void* d_out, int out_size, void* d_ws, size_t ws_size,
                              hipStream_t stream) {
  const float* x   = (const float*)d_in[0];  // (B,S,IN)
  const float* ts  = (const float*)d_in[1];  // (B,S)
  const float* Wi0 = (const float*)d_in[2];  // (17,3H)
  const float* bi0 = (const float*)d_in[3];  // (3H)
  const float* Wh0 = (const float*)d_in[4];  // (H,3H)
  const float* bh0 = (const float*)d_in[5];  // (3H)
  const float* WiR = (const float*)d_in[6];  // (2,H,3H)
  const float* biR = (const float*)d_in[7];  // (2,3H)
  const float* WhR = (const float*)d_in[8];  // (2,H,3H)
  const float* bhR = (const float*)d_in[9];  // (2,3H)
  float* out = (float*)d_out;

  char* ws = (char*)d_ws;
  size_t off = 0;
  __bf16* wti0 = (__bf16*)(ws + off); off += (size_t)H3_ * 32 * 2;
  __bf16* wth0 = (__bf16*)(ws + off); off += (size_t)H3_ * H_ * 2;
  __bf16* wti1 = (__bf16*)(ws + off); off += (size_t)H3_ * H_ * 2;
  __bf16* wti2 = (__bf16*)(ws + off); off += (size_t)H3_ * H_ * 2;
  __bf16* wth1 = (__bf16*)(ws + off); off += (size_t)H3_ * H_ * 2;
  __bf16* wth2 = (__bf16*)(ws + off); off += (size_t)H3_ * H_ * 2;
  float*  hf   = (float*)(ws + off);  off += (size_t)L_ * 2 * B_ * H_ * 4;
  __bf16* hb   = (__bf16*)(ws + off); off += (size_t)L_ * 2 * B_ * H_ * 2;
  __bf16* xin  = (__bf16*)(ws + off); off += (size_t)B_ * S_ * 32 * 2;
  (void)ws_size; (void)in_sizes; (void)n_in; (void)out_size;

  // Prep: transpose weights to col-major bf16; pack layer-0 input (per call,
  // deterministic).
  transpose_w<<<192, 256, 0, stream>>>(Wi0, wti0, IN_ + 1, H3_, 32);
  transpose_w<<<1024, 256, 0, stream>>>(Wh0, wth0, H_, H3_, H_);
  transpose_w<<<1024, 256, 0, stream>>>(WiR, wti1, H_, H3_, H_);
  transpose_w<<<1024, 256, 0, stream>>>(WiR + (size_t)H_ * H3_, wti2, H_, H3_, H_);
  transpose_w<<<1024, 256, 0, stream>>>(WhR, wth1, H_, H3_, H_);
  transpose_w<<<1024, 256, 0, stream>>>(WhR + (size_t)H_ * H3_, wth2, H_, H3_, H_);
  pack_xin<<<1024, 256, 0, stream>>>(x, ts, xin);
  zero_h<<<768, 256, 0, stream>>>(hf, hb, L_ * 2 * B_ * H_);

  const __bf16* WhT[3] = {wth0, wth1, wth2};
  const float*  biL[3] = {bi0, biR, biR + H3_};
  const float*  bhL[3] = {bh0, bhR, bhR + H3_};

  auto hF = [&](int l, int buf) { return hf + ((size_t)l * 2 + buf) * B_ * H_; };
  auto hB = [&](int l, int buf) { return hb + ((size_t)l * 2 + buf) * B_ * H_; };

  for (int t = 0; t < S_; ++t) {
    const int rp = t & 1, wp = rp ^ 1;
    // layer 0: input A = packed xin rows (stride S*32), K=32
    gru_step<32><<<64, 128, 0, stream>>>(
        xin + (size_t)t * 32, S_ * 32, wti0, biL[0], WhT[0], bhL[0],
        hB(0, rp), hF(0, rp), hB(0, wp), hF(0, wp), (float*)nullptr, t);
    gru_step<H_><<<64, 128, 0, stream>>>(
        hB(0, wp), H_, wti1, biL[1], WhT[1], bhL[1],
        hB(1, rp), hF(1, rp), hB(1, wp), hF(1, wp), (float*)nullptr, t);
    gru_step<H_><<<64, 128, 0, stream>>>(
        hB(1, wp), H_, wti2, biL[2], WhT[2], bhL[2],
        hB(2, rp), hF(2, rp), hB(2, wp), hF(2, wp), out, t);
  }
  // After t = S-1 (odd), the write buffer was 0 -> final states live in buf 0.
  copy_final<<<384, 256, 0, stream>>>(hf, out + (size_t)B_ * S_ * H_);
}
